// RecommenderNet_14139032339110
// MI455X (gfx1250) — matrix operand order repaired and checked by reading, not compiled, not run on hardware
//
#include <hip/hip_runtime.h>

// ---------------- problem constants (match reference) ----------------
#define U_    100000
#define I_    50000
#define DD    64
#define G_    4
#define L_    2
#define P_    50
#define BB    4096
#define EE    800000
#define HH    4
#define ATTL  2
#define SSLW  32
#define NSS_  8192
#define CD    192       // (L_+1)*DD
#define SLOPE 0.5f

typedef __attribute__((ext_vector_type(16))) __bf16 v16bf;
typedef __attribute__((ext_vector_type(8)))  float  v8f;
typedef __attribute__((ext_vector_type(4)))  unsigned int u32x4;
typedef __attribute__((ext_vector_type(8)))  int  i32x8;
typedef __attribute__((ext_vector_type(4)))  int  i32x4;

__device__ __forceinline__ float lrelu(float x){ return x >= 0.f ? x : SLOPE * x; }
__device__ __forceinline__ float sigm (float x){ return 1.f / (1.f + __expf(-x)); }
__device__ __forceinline__ int   iclamp(int v,int lo,int hi){ return v<lo?lo:(v>hi?hi:v); }

// =====================================================================
// Row-parallel GEMM on the WMMA pipe, W tile staged in LDS by the TDM:
//   C[rows x N] = act( A[rows x K] @ W + bias ) (+ resid)
// - A: f32 row-major (lda), loaded as 2x2 float4 (global_load_b128) per
//   lane per 32-K chunk, converted to bf16 in-register.
// - W: f32, (K x N, ldw) or transposed (N x K); the 64-column tile this
//   block needs (<= 192x64 f32 = 48KB) is DMA'd global->LDS once via
//   tensor_load_to_lds (TENSORcnt + s_wait_tensorcnt), then read via ds.
// - f32 accumulate via v_wmma_f32_16x16x32_bf16 (wave32).
// Block = 256 thr = 8 waves = 32x64 C tile. N % 64 == 0, K % 32 == 0.
// Compile-time WT/ACT/HB/HR keep the inner loop branch-free.
// =====================================================================
template<int WT, int ACT, bool HB, bool HR>
__global__ __launch_bounds__(256)
void gemm_wmma(const float* __restrict__ A, int lda,
               const float* __restrict__ W, int ldw,
               const float* __restrict__ bias,
               const float* __restrict__ resid,
               float* __restrict__ C, int ldc, int coff,
               int rows, int K)
{
    __shared__ float wlds[CD * DD];            // 48 KB (max K=192, 64 cols)

    const int lane  = threadIdx.x & 31;
    const int wave  = threadIdx.x >> 5;
    const int mwave = wave >> 2;               // 0..1
    const int nwave = wave & 3;                // 0..3
    const int row0  = blockIdx.x * 32 + mwave * 16;
    const int colB0 = blockIdx.y * 64;
    const int m     = lane & 15;
    const int hi    = lane >> 4;
    const int rowA  = min(row0 + m, rows - 1); // clamp for ragged tail
    const int nl    = nwave * 16 + m;          // local col 0..63
    const int n     = colB0 + nl;              // global col

    // ---- TDM: DMA this block's W tile into LDS (wave 0 issues) ----
    if (wave == 0) {
        unsigned long long gaddr;
        unsigned int td0, td1;
        const unsigned long long stride = (unsigned long long)ldw; // dim0 stride (elems)
        if (WT) { gaddr = (unsigned long long)(size_t)(const void*)(W + (size_t)colB0 * ldw);
                  td0 = (unsigned)K; td1 = 64u; }                  // 64 rows of K contig
        else    { gaddr = (unsigned long long)(size_t)(const void*)(W + colB0);
                  td0 = 64u; td1 = (unsigned)K; }                  // K rows of 64 contig
        const unsigned int ldsa = (unsigned int)(size_t)(void*)&wlds[0];
        u32x4 g0;
        g0[0] = 1u;                                            // count=1, flags 0
        g0[1] = ldsa;                                          // lds_addr (bytes)
        g0[2] = (unsigned)(gaddr & 0xffffffffull);             // global_addr[31:0]
        g0[3] = (unsigned)((gaddr >> 32) & 0x1ffffffull)       // global_addr[56:32]
              | (2u << 30);                                    // type = 2 ("image")
        i32x8 g1;
        g1[0] = (int)(2u << 16);                               // data_size = 4B
        g1[1] = (int)((td0 & 0xffffu) << 16);                  // tensor_dim0 lo16
        g1[2] = (int)(((td0 >> 16) & 0xffffu) | ((td1 & 0xffffu) << 16)); // d0 hi / d1 lo
        g1[3] = (int)(((td1 >> 16) & 0xffffu) | ((td0 & 0xffffu) << 16)); // d1 hi / tile_dim0
        g1[4] = (int)(td1 & 0xffffu);                          // tile_dim1 (tile_dim2=0)
        g1[5] = (int)(stride & 0xffffffffull);                 // tensor_dim0_stride lo32
        g1[6] = (int)((stride >> 32) & 0xffffull);             // stride hi16, dim1_stride=0
        g1[7] = 0;
        i32x4 z4 = {0, 0, 0, 0};
#if defined(__clang_major__) && (__clang_major__ >= 23)
        i32x8 z8 = {0, 0, 0, 0, 0, 0, 0, 0};
        __builtin_amdgcn_tensor_load_to_lds(g0, g1, z4, z4, z8, 0);
#else
        __builtin_amdgcn_tensor_load_to_lds(g0, g1, z4, z4, 0);
#endif
        __builtin_amdgcn_s_wait_tensorcnt(0);
    }
    __syncthreads();

    v8f acc = {};
    const float* arow = A + (size_t)rowA * lda;
    for (int k0 = 0; k0 < K; k0 += 32) {
        // A lane layout (16-bit 16x32): this lane needs K = k0+hi*8+{0..7} and +16..23
        const float4* ap0 = (const float4*)(arow + k0 + hi * 8);
        const float4* ap1 = (const float4*)(arow + k0 + hi * 8 + 16);
        const float4 a0 = ap0[0], a1 = ap0[1];
        const float4 a2 = ap1[0], a3 = ap1[1];
        v16bf a, b;
        a[0]=(__bf16)a0.x;  a[1]=(__bf16)a0.y;  a[2]=(__bf16)a0.z;  a[3]=(__bf16)a0.w;
        a[4]=(__bf16)a1.x;  a[5]=(__bf16)a1.y;  a[6]=(__bf16)a1.z;  a[7]=(__bf16)a1.w;
        a[8]=(__bf16)a2.x;  a[9]=(__bf16)a2.y;  a[10]=(__bf16)a2.z; a[11]=(__bf16)a2.w;
        a[12]=(__bf16)a3.x; a[13]=(__bf16)a3.y; a[14]=(__bf16)a3.z; a[15]=(__bf16)a3.w;
#pragma unroll
        for (int j = 0; j < 16; ++j) {
            const int kB = k0 + (j < 8 ? hi * 8 + j : 16 + hi * 8 + (j - 8));   // B layout 32x16: K = hi*8+{0..7}, 16+hi*8+{0..7} (matches A)
            b[j] = (__bf16)(WT ? wlds[nl * K + kB] : wlds[kB * DD + nl]);
        }
        acc = __builtin_amdgcn_wmma_f32_16x16x32_bf16(
                  false, a, false, b, (short)0, acc, false, false);
    }
    const float bv = HB ? bias[n] : 0.f;
#pragma unroll
    for (int r = 0; r < 8; ++r) {
        const int row = row0 + r + hi * 8;     // C/D: vgpr r -> M=r (+8 for hi half)
        if (row < rows) {
            float v = acc[r] + bv;
            if (ACT == 1) v = lrelu(v);
            const size_t off = (size_t)row * ldc + coff + n;
            if (HR) v += resid[off];
            C[off] = v;
        }
    }
}

// ---------------- small utility kernels ----------------
__global__ void zero_buf(float* p, long n){
    long i = (long)blockIdx.x * blockDim.x + threadIdx.x;
    if (i < n) p[i] = 0.f;
}
__global__ void set_scalar(float* p, float v){
    if (threadIdx.x == 0 && blockIdx.x == 0) *p = v;
}
__global__ void zero_col(float* buf, int stride, int off, int rows){
    int i = blockIdx.x * blockDim.x + threadIdx.x;
    if (i >= rows * DD) return;
    buf[(size_t)(i >> 6) * stride + off + (i & 63)] = 0.f;
}
__global__ void copy_col(const float* __restrict__ src, float* __restrict__ dst,
                         int rows, int stride, int off){
    int i = blockIdx.x * blockDim.x + threadIdx.x;
    if (i >= rows * DD) return;
    dst[(size_t)(i >> 6) * stride + off + (i & 63)] = src[i];
}

// ---------------- graph propagation (segment_sum) ----------------
// one block per edge, 64 lanes = feature dim; coalesced gather + atomic scatter
__global__ void prop_scatter(const float* __restrict__ src, int srcStride, int srcOff,
                             float* __restrict__ dst, int dstStride, int dstOff,
                             const int* __restrict__ idx, int nSrc, int nTgt)
{
    const int e = blockIdx.x;
    const int d = threadIdx.x;           // 0..63
    const int t = iclamp(idx[2*e],     0, nTgt - 1);
    const int s = iclamp(idx[2*e + 1], 0, nSrc - 1);
    const float v = src[(size_t)s * srcStride + srcOff + d];
    atomicAdd(&dst[(size_t)t * dstStride + dstOff + d], v);
}
// next = lrelu(agg) + prev
__global__ void prop_finalize(float* buf, int stride, int prevOff, int curOff, int rows){
    int i = blockIdx.x * blockDim.x + threadIdx.x;
    if (i >= rows * DD) return;
    const int r = i >> 6, d = i & 63;
    float* cur = &buf[(size_t)r * stride + curOff + d];
    *cur = lrelu(*cur) + buf[(size_t)r * stride + prevOff + d];
}

// ---------------- GRU pointwise step ----------------
__global__ void gru_pointwise(const float* __restrict__ gi, const float* __restrict__ gh,
                              float* __restrict__ h, float* __restrict__ outSeq,
                              int rows, int t)
{
    int i = blockIdx.x * blockDim.x + threadIdx.x;
    if (i >= rows * DD) return;
    const int row = i >> 6, d = i & 63;
    const size_t g = (size_t)row * CD + d;
    const float r  = sigm(gi[g]       + gh[g]);
    const float z  = sigm(gi[g + 64]  + gh[g + 64]);
    const float nn = tanhf(gi[g + 128] + r * gh[g + 128]);
    const float hv = (1.f - z) * nn + z * h[i];
    h[i] = hv;
    outSeq[(size_t)row * (G_ * DD) + t * DD + d] = hv;
}

// ---------------- layernorm over last dim (64) ----------------
__global__ void layernorm_rows(const float* __restrict__ x, float* __restrict__ y,
                               const float* __restrict__ g, const float* __restrict__ b,
                               int rows)
{
    int row = blockIdx.x * blockDim.x + threadIdx.x;
    if (row >= rows) return;
    const float* xr = x + (size_t)row * DD;
    float m = 0.f;
    for (int d = 0; d < DD; ++d) m += xr[d];
    m *= (1.f / DD);
    float v = 0.f;
    for (int d = 0; d < DD; ++d){ float t = xr[d] - m; v += t * t; }
    const float rs = rsqrtf(v * (1.f / DD) + 1e-5f);
    float* yr = y + (size_t)row * DD;
    for (int d = 0; d < DD; ++d) yr[d] = (xr[d] - m) * rs * g[d] + b[d];
}

// ---------------- tiny MHA over seq-len G=4, folded sum over queries -------
__global__ void mha_small_att(const float* __restrict__ q, const float* __restrict__ k,
                              const float* __restrict__ v, float* __restrict__ osum, int N)
{
    int i = blockIdx.x * blockDim.x + threadIdx.x;
    if (i >= N * HH) return;
    const int n = i / HH, h = i % HH;
    float out[16];
#pragma unroll
    for (int j = 0; j < 16; ++j) out[j] = 0.f;
    for (int qt = 0; qt < G_; ++qt) {
        const float* qr = q + ((size_t)n * G_ + qt) * DD + h * 16;
        float qv[16];
#pragma unroll
        for (int j = 0; j < 16; ++j) qv[j] = qr[j];
        float sc[G_], mx = -3.0e38f;
        for (int kt = 0; kt < G_; ++kt) {
            const float* kr = k + ((size_t)n * G_ + kt) * DD + h * 16;
            float s = 0.f;
#pragma unroll
            for (int j = 0; j < 16; ++j) s += qv[j] * kr[j];
            s *= 0.25f;                     // 1/sqrt(16)
            sc[kt] = s; mx = fmaxf(mx, s);
        }
        float den = 0.f;
        for (int kt = 0; kt < G_; ++kt){ sc[kt] = __expf(sc[kt] - mx); den += sc[kt]; }
        const float inv = 1.f / den;
        for (int kt = 0; kt < G_; ++kt) {
            const float* vr = v + ((size_t)n * G_ + kt) * DD + h * 16;
            const float w = sc[kt] * inv;
#pragma unroll
            for (int j = 0; j < 16; ++j) out[j] += w * vr[j];
        }
    }
    float* o = osum + (size_t)n * DD + h * 16;
#pragma unroll
    for (int j = 0; j < 16; ++j) o[j] = out[j];
}

// ---------------- sequence attention (P=50), two-pass softmax ----------------
__global__ void seq_att(const float* __restrict__ q, const float* __restrict__ k,
                        const float* __restrict__ v, float* __restrict__ o)
{
    int i = blockIdx.x * blockDim.x + threadIdx.x;
    if (i >= BB * HH * P_) return;
    const int b  = i / (HH * P_);
    const int h  = (i / P_) % HH;
    const int qt = i % P_;
    const float* qr = q + ((size_t)b * P_ + qt) * DD + h * 16;
    float qv[16];
#pragma unroll
    for (int j = 0; j < 16; ++j) qv[j] = qr[j];
    float mx = -3.0e38f;
    for (int kt = 0; kt < P_; ++kt) {
        const float* kr = k + ((size_t)b * P_ + kt) * DD + h * 16;
        float s = 0.f;
#pragma unroll
        for (int j = 0; j < 16; ++j) s += qv[j] * kr[j];
        mx = fmaxf(mx, s * 0.25f);
    }
    float den = 0.f, accv[16];
#pragma unroll
    for (int j = 0; j < 16; ++j) accv[j] = 0.f;
    for (int kt = 0; kt < P_; ++kt) {
        const float* kr = k + ((size_t)b * P_ + kt) * DD + h * 16;
        float s = 0.f;
#pragma unroll
        for (int j = 0; j < 16; ++j) s += qv[j] * kr[j];
        const float e = __expf(s * 0.25f - mx);
        den += e;
        const float* vr = v + ((size_t)b * P_ + kt) * DD + h * 16;
#pragma unroll
        for (int j = 0; j < 16; ++j) accv[j] += e * vr[j];
    }
    const float inv = 1.f / den;
    float* orow = o + ((size_t)b * P_ + qt) * DD + h * 16;
#pragma unroll
    for (int j = 0; j < 16; ++j) orow[j] = accv[j] * inv;
}

// ---------------- build x0 = LN(mask*(fi[seq] + pos)) ----------------
__global__ void build_seq_x0(const float* __restrict__ fi, const float* __restrict__ pos,
                             const float* __restrict__ mask, const int* __restrict__ seq,
                             float* __restrict__ x,
                             const float* __restrict__ g, const float* __restrict__ b)
{
    int row = blockIdx.x * blockDim.x + threadIdx.x;
    if (row >= BB * P_) return;
    const int p  = row % P_;
    const int it = iclamp(seq[row], 0, I_ - 1);
    const float mk = mask[row];
    float buf[DD];
    float m = 0.f;
    for (int d = 0; d < DD; ++d) {
        buf[d] = mk * (fi[(size_t)it * DD + d] + pos[(size_t)p * DD + d]);
        m += buf[d];
    }
    m *= (1.f / DD);
    float v = 0.f;
    for (int d = 0; d < DD; ++d){ float t = buf[d] - m; v += t * t; }
    const float rs = rsqrtf(v * (1.f / DD) + 1e-5f);
    float* xr = x + (size_t)row * DD;
    for (int d = 0; d < DD; ++d) xr[d] = (buf[d] - m) * rs * g[d] + b[d];
}

__global__ void sum_over_p(const float* __restrict__ x, float* __restrict__ attu){
    int i = blockIdx.x * blockDim.x + threadIdx.x;
    if (i >= BB * DD) return;
    const int b = i >> 6, d = i & 63;
    float s = 0.f;
    for (int p = 0; p < P_; ++p) s += x[((size_t)b * P_ + p) * DD + d];
    attu[i] = s;
}

__global__ void preds_kernel(const float* __restrict__ fu, const float* __restrict__ fi,
                             const float* __restrict__ attu,
                             const int* __restrict__ uids, const int* __restrict__ iids,
                             const int* __restrict__ ulocs, float* __restrict__ out)
{
    int b = blockIdx.x * blockDim.x + threadIdx.x;
    if (b >= BB) return;
    const int u  = iclamp(uids[b], 0, U_ - 1);
    const int it = iclamp(iids[b], 0, I_ - 1);
    const int ul = iclamp(ulocs[b], 0, BB - 1);
    float s = 0.f;
    for (int d = 0; d < DD; ++d)
        s += (fu[(size_t)u * DD + d] + attu[(size_t)ul * DD + d]) * fi[(size_t)it * DD + d];
    out[b] = s;
}

// ---------------- meta-MLP user weights ----------------
__global__ void meta_kernel(const float* __restrict__ fu, const float* __restrict__ uvbuf,
                            const float* __restrict__ m1w, const float* __restrict__ m1b,
                            const float* __restrict__ m2w, const float* __restrict__ m2b,
                            float* __restrict__ uw)
{
    int i = blockIdx.x * blockDim.x + threadIdx.x;
    if (i >= G_ * U_) return;
    const int g = i / U_, u = i % U_;
    const float* f = fu    + (size_t)u * DD;
    const float* e = uvbuf + (size_t)u * (G_ * DD) + g * DD;
    float zb[DD];
    for (int d = 0; d < DD; ++d) zb[d] = f[d] + e[d] + f[d] * e[d];
    float acc2 = m2b[0];
    for (int j = 0; j < SSLW; ++j) {
        float s = m1b[j];
        for (int d = 0; d < DD; ++d) s += zb[d] * m1w[(size_t)d * SSLW + j];
        acc2 += lrelu(s) * m2w[j];
    }
    uw[(size_t)g * U_ + u] = sigm(acc2);
}

// ---------------- SSL stage 1: s_fin and p1 (preds_one) ----------------
__global__ void ssl_p1(const float* __restrict__ fu, const float* __restrict__ fi,
                       const float* __restrict__ uvbuf, const float* __restrict__ ivbuf,
                       const int* __restrict__ suids, const int* __restrict__ siids,
                       float* __restrict__ sfin, float* __restrict__ predsOne)
{
    int i = blockIdx.x * blockDim.x + threadIdx.x;
    if (i >= G_ * NSS_) return;
    const int g = i / NSS_;
    const int su = iclamp(suids[i], 0, U_ - 1);
    const int si = iclamp(siids[i], 0, I_ - 1);
    const float* a = fu + (size_t)su * DD;
    const float* b = fi + (size_t)si * DD;
    const float* c = uvbuf + (size_t)su * (G_ * DD) + g * DD;
    const float* d = ivbuf + (size_t)si * (G_ * DD) + g * DD;
    float s1 = 0.f, s2 = 0.f;
    for (int t = 0; t < DD; ++t) {
        s1 += lrelu(a[t] * b[t]);
        s2 += lrelu(c[t] * d[t]);
    }
    sfin[i]     = s1;
    predsOne[i] = s2;
}

// ---------------- SSL stage 2: hinge sum ----------------
__global__ void ssl_loss(const float* __restrict__ sfin, const float* __restrict__ predsOne,
                         const float* __restrict__ uw, const int* __restrict__ suids,
                         float* __restrict__ loss)
{
    const int samp = NSS_ / 2;
    int i = blockIdx.x * blockDim.x + threadIdx.x;
    if (i >= G_ * samp) return;
    const int g = i / samp, j = i % samp;
    const int suP = iclamp(suids[g * NSS_ + j],        0, U_ - 1);
    const int suN = iclamp(suids[g * NSS_ + samp + j], 0, U_ - 1);
    const float sf = uw[(size_t)g * U_ + suP] * sfin[g * NSS_ + j]
                   - uw[(size_t)g * U_ + suN] * sfin[g * NSS_ + samp + j];
    const float dp = predsOne[g * NSS_ + j] - predsOne[g * NSS_ + samp + j];
    const float t  = 1.f - sf * dp;
    if (t > 0.f) atomicAdd(loss, t);
}

// =====================================================================
// host-side orchestration
// =====================================================================
static inline void launch_gemm(hipStream_t s, const float* A, int lda,
                               const float* W, int ldw, int wT,
                               const float* bias, const float* resid,
                               float* C, int ldc, int coff,
                               int rows, int K, int N, int act)
{
    dim3 g((rows + 31) / 32, N / 64);
    if (wT)        gemm_wmma<1,0,true ,false><<<g,256,0,s>>>(A,lda,W,ldw,bias,resid,C,ldc,coff,rows,K);
    else if (act)  gemm_wmma<0,1,false,true ><<<g,256,0,s>>>(A,lda,W,ldw,bias,resid,C,ldc,coff,rows,K);
    else if (bias) gemm_wmma<0,0,true ,false><<<g,256,0,s>>>(A,lda,W,ldw,bias,resid,C,ldc,coff,rows,K);
    else           gemm_wmma<0,0,false,false><<<g,256,0,s>>>(A,lda,W,ldw,bias,resid,C,ldc,coff,rows,K);
}
static inline int nb(long n){ return (int)((n + 255) / 256); }

extern "C" void kernel_launch(void* const* d_in, const int* in_sizes, int n_in,
                              void* d_out, int out_size, void* d_ws, size_t ws_size,
                              hipStream_t stream)
{
    (void)in_sizes; (void)n_in; (void)out_size; (void)ws_size;
    // ---- inputs (setup_inputs order) ----
    const float* u_embed = (const float*)d_in[0];
    const float* i_embed = (const float*)d_in[1];
    const float* pos     = (const float*)d_in[2];
    const float* w_su    = (const float*)d_in[3];
    const float* b_su    = (const float*)d_in[4];
    const float* w_si    = (const float*)d_in[5];
    const float* b_si    = (const float*)d_in[6];
    const float* gru_wi  = (const float*)d_in[7];
    const float* gru_wh  = (const float*)d_in[8];
    const float* gru_bi  = (const float*)d_in[9];
    const float* gru_bh  = (const float*)d_in[10];
    const float* au_wq   = (const float*)d_in[11];
    const float* au_wk   = (const float*)d_in[12];
    const float* au_wv   = (const float*)d_in[13];
    const float* au_wo   = (const float*)d_in[14];
    const float* ai_wq   = (const float*)d_in[15];
    const float* ai_wk   = (const float*)d_in[16];
    const float* ai_wv   = (const float*)d_in[17];
    const float* ai_wo   = (const float*)d_in[18];
    const float* sq_wq   = (const float*)d_in[19];
    const float* sq_wk   = (const float*)d_in[20];
    const float* sq_wv   = (const float*)d_in[21];
    const float* sq_wo   = (const float*)d_in[22];
    const float* ln_g    = (const float*)d_in[23];
    const float* ln_b    = (const float*)d_in[24];
    const float* m1_w    = (const float*)d_in[25];
    const float* m1_b    = (const float*)d_in[26];
    const float* m2_w    = (const float*)d_in[27];
    const float* m2_b    = (const float*)d_in[28];
    const float* mask    = (const float*)d_in[29];
    const int*   uids    = (const int*)d_in[30];
    const int*   iids    = (const int*)d_in[31];
    const int*   seq     = (const int*)d_in[32];
    const int*   ulocs   = (const int*)d_in[33];
    const int*   suids   = (const int*)d_in[34];
    const int*   siids   = (const int*)d_in[35];
    const int*   adj     = (const int*)d_in[36];
    const int*   adj_tp  = (const int*)d_in[37];
    float* out = (float*)d_out;            // [preds(B) | loss(1) | preds_one(G*NSS)]

    // ---- workspace layout (floats) ----
    float* ws = (float*)d_ws;
    size_t off = 0;
    auto alloc = [&](size_t n){ float* p = ws + off; off += n; return p; };
    float* uvbuf  = alloc((size_t)U_ * G_ * DD);   // user_vectors stacked (U,G,64)
    float* ivbuf  = alloc((size_t)I_ * G_ * DD);
    float* uvr    = alloc((size_t)U_ * G_ * DD);   // GRU out; later reused as xseq
    float* ivr    = alloc((size_t)I_ * G_ * DD);
    float* conc_u = alloc((size_t)U_ * CD);        // concat embeds; later gi; later seq-o
    float* conc_i = alloc((size_t)I_ * CD);
    float* ghb    = alloc((size_t)U_ * CD);
    float* hbuf   = alloc((size_t)U_ * DD);
    float* xln    = alloc((size_t)U_ * G_ * DD);
    float* qb     = alloc((size_t)U_ * G_ * DD);
    float* kb     = alloc((size_t)U_ * G_ * DD);
    float* vb     = alloc((size_t)U_ * G_ * DD);
    float* osum   = alloc((size_t)U_ * DD);
    float* fub    = alloc((size_t)U_ * DD);
    float* fib    = alloc((size_t)I_ * DD);
    float* attu   = alloc((size_t)BB * DD);
    float* uw     = alloc((size_t)G_ * U_);
    float* sfin   = alloc((size_t)G_ * NSS_);

    // ================= 1) graph propagation + feature transform =================
    for (int k = 0; k < G_; ++k) {
        copy_col<<<nb((long)U_ * DD), 256, 0, stream>>>(u_embed + (size_t)k * U_ * DD, conc_u, U_, CD, 0);
        copy_col<<<nb((long)I_ * DD), 256, 0, stream>>>(i_embed + (size_t)k * I_ * DD, conc_i, I_, CD, 0);
        for (int j = 0; j < L_; ++j) {
            zero_col<<<nb((long)U_ * DD), 256, 0, stream>>>(conc_u, CD, (j + 1) * DD, U_);
            zero_col<<<nb((long)I_ * DD), 256, 0, stream>>>(conc_i, CD, (j + 1) * DD, I_);
            prop_scatter<<<EE, DD, 0, stream>>>(conc_i, CD, j * DD, conc_u, CD, (j + 1) * DD,
                                                adj    + (size_t)k * EE * 2, I_, U_);
            prop_scatter<<<EE, DD, 0, stream>>>(conc_u, CD, j * DD, conc_i, CD, (j + 1) * DD,
                                                adj_tp + (size_t)k * EE * 2, U_, I_);
            prop_finalize<<<nb((long)U_ * DD), 256, 0, stream>>>(conc_u, CD, j * DD, (j + 1) * DD, U_);
            prop_finalize<<<nb((long)I_ * DD), 256, 0, stream>>>(conc_i, CD, j * DD, (j + 1) * DD, I_);
        }
        launch_gemm(stream, conc_u, CD, w_su, DD, 0, b_su, nullptr, uvbuf, G_ * DD, k * DD, U_, CD, DD, 0);
        launch_gemm(stream, conc_i, CD, w_si, DD, 0, b_si, nullptr, ivbuf, G_ * DD, k * DD, I_, CD, DD, 0);
    }

    // ================= 2) GRU over the G=4 graph axis =================
    float* gib = conc_u;  // reuse (U*192)
    zero_buf<<<nb((long)U_ * DD), 256, 0, stream>>>(hbuf, (long)U_ * DD);
    for (int t = 0; t < G_; ++t) {
        launch_gemm(stream, uvbuf + t * DD, G_ * DD, gru_wi, DD, 1, gru_bi, nullptr, gib, CD, 0, U_, DD, CD, 0);
        launch_gemm(stream, hbuf, DD,       gru_wh, DD, 1, gru_bh, nullptr, ghb, CD, 0, U_, DD, CD, 0);
        gru_pointwise<<<nb((long)U_ * DD), 256, 0, stream>>>(gib, ghb, hbuf, uvr, U_, t);
    }
    zero_buf<<<nb((long)I_ * DD), 256, 0, stream>>>(hbuf, (long)I_ * DD);
    for (int t = 0; t < G_; ++t) {
        launch_gemm(stream, ivbuf + t * DD, G_ * DD, gru_wi, DD, 1, gru_bi, nullptr, gib, CD, 0, I_, DD, CD, 0);
        launch_gemm(stream, hbuf, DD,       gru_wh, DD, 1, gru_bh, nullptr, ghb, CD, 0, I_, DD, CD, 0);
        gru_pointwise<<<nb((long)I_ * DD), 256, 0, stream>>>(gib, ghb, hbuf, ivr, I_, t);
    }

    // ================= 3) fu / fi : LN -> MHA(s=4) summed -> @wo =================
    layernorm_rows<<<nb((long)U_ * G_), 256, 0, stream>>>(uvr, xln, ln_g, ln_b, U_ * G_);
    launch_gemm(stream, xln, DD, au_wq, DD, 0, nullptr, nullptr, qb, DD, 0, U_ * G_, DD, DD, 0);
    launch_gemm(stream, xln, DD, au_wk, DD, 0, nullptr, nullptr, kb, DD, 0, U_ * G_, DD, DD, 0);
    launch_gemm(stream, xln, DD, au_wv, DD, 0, nullptr, nullptr, vb, DD, 0, U_ * G_, DD, DD, 0);
    mha_small_att<<<nb((long)U_ * HH), 256, 0, stream>>>(qb, kb, vb, osum, U_);
    launch_gemm(stream, osum, DD, au_wo, DD, 0, nullptr, nullptr, fub, DD, 0, U_, DD, DD, 0);

    layernorm_rows<<<nb((long)I_ * G_), 256, 0, stream>>>(ivr, xln, ln_g, ln_b, I_ * G_);
    launch_gemm(stream, xln, DD, ai_wq, DD, 0, nullptr, nullptr, qb, DD, 0, I_ * G_, DD, DD, 0);
    launch_gemm(stream, xln, DD, ai_wk, DD, 0, nullptr, nullptr, kb, DD, 0, I_ * G_, DD, DD, 0);
    launch_gemm(stream, xln, DD, ai_wv, DD, 0, nullptr, nullptr, vb, DD, 0, I_ * G_, DD, DD, 0);
    mha_small_att<<<nb((long)I_ * HH), 256, 0, stream>>>(qb, kb, vb, osum, I_);
    launch_gemm(stream, osum, DD, ai_wo, DD, 0, nullptr, nullptr, fib, DD, 0, I_, DD, DD, 0);

    // ================= 4) sequence transformer (B=4096, P=50) =================
    float* xseq = uvr;     // reuse (B*P*64 <= U*G*64)
    float* oseq = conc_u;  // reuse (B*P*64 <= U*192)
    build_seq_x0<<<nb((long)BB * P_), 256, 0, stream>>>(fib, pos, mask, seq, xseq, ln_g, ln_b);
    for (int l = 0; l < ATTL; ++l) {
        const size_t wo = (size_t)l * DD * DD;
        layernorm_rows<<<nb((long)BB * P_), 256, 0, stream>>>(xseq, xln, ln_g, ln_b, BB * P_);
        launch_gemm(stream, xln, DD, sq_wq + wo, DD, 0, nullptr, nullptr, qb, DD, 0, BB * P_, DD, DD, 0);
        launch_gemm(stream, xln, DD, sq_wk + wo, DD, 0, nullptr, nullptr, kb, DD, 0, BB * P_, DD, DD, 0);
        launch_gemm(stream, xln, DD, sq_wv + wo, DD, 0, nullptr, nullptr, vb, DD, 0, BB * P_, DD, DD, 0);
        seq_att<<<nb((long)BB * HH * P_), 256, 0, stream>>>(qb, kb, vb, oseq);
        // x = lrelu(o @ wo) + x   (fused activation + residual)
        launch_gemm(stream, oseq, DD, sq_wo + wo, DD, 0, nullptr, xseq, xseq, DD, 0, BB * P_, DD, DD, 1);
    }
    sum_over_p<<<nb((long)BB * DD), 256, 0, stream>>>(xseq, attu);

    // ================= 5) predictions / meta / SSL =================
    preds_kernel<<<nb((long)BB), 256, 0, stream>>>(fub, fib, attu, uids, iids, ulocs, out);
    meta_kernel<<<nb((long)G_ * U_), 256, 0, stream>>>(fub, uvbuf, m1_w, m1_b, m2_w, m2_b, uw);
    ssl_p1<<<nb((long)G_ * NSS_), 256, 0, stream>>>(fub, fib, uvbuf, ivbuf, suids, siids,
                                                    sfin, out + BB + 1);
    set_scalar<<<1, 1, 0, stream>>>(out + BB, 0.f);
    ssl_loss<<<nb((long)G_ * (NSS_ / 2)), 256, 0, stream>>>(sfin, out + BB + 1, uw, suids, out + BB);
}
